// PatchRefiner_84146999263749
// MI455X (gfx1250) — matrix operand order repaired and checked by reading, not compiled, not run on hardware
//
#include <hip/hip_runtime.h>
#include <hip/hip_bf16.h>

typedef __attribute__((ext_vector_type(16))) _Float16 v16h;
typedef __attribute__((ext_vector_type(8)))  float    v8f;

__device__ __forceinline__ v8f wmma_f16(v16h a, v16h b, v8f c) {
    return __builtin_amdgcn_wmma_f32_16x16x32_f16(false, a, false, b, (short)0, c,
                                                  false, false);
}

// A-fragment (16x32 f16, M x K) per ISA 7.12.2:
// lane l: m = m0 + (l&15), kb = k0 + (l>>4)*8; halves K = kb..kb+7 and kb+16..kb+23
__device__ __forceinline__ v16h load_a_frag(const _Float16* p, int lda, int m0, int k0) {
    int l  = threadIdx.x & 31;
    int m  = m0 + (l & 15);
    int kb = k0 + ((l >> 4) << 3);
    const _Float16* p0 = p + (size_t)m * lda + kb;
    v16h r;
#pragma unroll
    for (int i = 0; i < 8; ++i) { r[i] = p0[i]; r[8 + i] = p0[16 + i]; }
    return r;
}

// B-fragment (32x16 f16, K x N), weight stored row-major [n][K]:
// lane l: n = n0 + (l&15), kb = k0 + (l>>4)*8; halves K = kb..kb+7 and kb+16..kb+23
__device__ __forceinline__ v16h load_b_frag(const _Float16* w, int ldk, int n0, int k0) {
    int l  = threadIdx.x & 31;
    int n  = n0 + (l & 15);
    int kb = k0 + ((l >> 4) << 3);
    const _Float16* p0 = w + (size_t)n * ldk + kb;
    v16h r;
#pragma unroll
    for (int i = 0; i < 8; ++i) { r[i] = p0[i]; r[8 + i] = p0[16 + i]; }
    return r;
}

// CDNA5 async global->LDS copy (16 bytes), tracked by ASYNCcnt.
// LDS destination address = low 32 bits of the generic shared pointer (ISA 10.2).
__device__ __forceinline__ void async_copy_b128(void* lds_dst, const void* gsrc) {
    uint32_t la = (uint32_t)(uintptr_t)lds_dst;
    uint64_t ga = (uint64_t)(uintptr_t)gsrc;
    asm volatile("global_load_async_to_lds_b128 %0, %1, off"
                 :: "v"(la), "v"(ga) : "memory");
}
__device__ __forceinline__ void wait_asynccnt0() {
    asm volatile("s_wait_asynccnt 0x0" ::: "memory");
}

// ---------------- weight f32 -> f16 convert ----------------
__global__ __launch_bounds__(256) void cvt_f16_kernel(const float* __restrict__ s,
                                                      _Float16* __restrict__ d, int n) {
    int i = blockIdx.x * 256 + threadIdx.x;
    if (i < n) d[i] = (_Float16)s[i];
}

// ---------------- stage 1: fused pointnet + maxpool ----------------
// grid = 1024 (one block per (b,a) patch), block = 256 (8 waves)
__global__ __launch_bounds__(256)
void pointnet_wmma(const float* __restrict__ regions, const float* __restrict__ anchors,
                   const float* __restrict__ W1, const float* __restrict__ b1,
                   const float* __restrict__ W2, const float* __restrict__ b2,
                   float* __restrict__ feat) {
    __shared__ _Float16 sh1[128][136];   // h1 chunk: 128 points x 128 ch (padded)
    __shared__ _Float16 sW2[256 * 128];  // full W2 in f16
    __shared__ float    sW1[384];
    __shared__ float    sb1[128];
    __shared__ float    cen[4];

    const int tid = threadIdx.x, lane = tid & 31, wave = tid >> 5;
    const int pa  = blockIdx.x;

    for (int i = tid; i < 256 * 128; i += 256) sW2[i] = (_Float16)W2[i];
    for (int i = tid; i < 384; i += 256) sW1[i] = W1[i];
    for (int i = tid; i < 128; i += 256) sb1[i] = b1[i];
    if (tid < 3) cen[tid] = anchors[pa * 3 + tid];

    const float* reg = regions + (size_t)pa * 3072;  // [3][1024]
    const int n0a = wave * 32, n0b = n0a + 16;
    float rmax0 = -3.0e38f, rmax1 = -3.0e38f;

    for (int ch = 0; ch < 8; ++ch) {
        __syncthreads();
        // layer1: 2 threads per point, 64 channels each
        {
            const int pt = tid >> 1, cof = (tid & 1) << 6;
            const int gp = (ch << 7) + pt;
            const float x = reg[gp] - cen[0];
            const float y = reg[1024 + gp] - cen[1];
            const float z = reg[2048 + gp] - cen[2];
#pragma unroll 8
            for (int j = 0; j < 64; ++j) {
                const int c = cof + j;
                float h = fmaf(sW1[c * 3], x,
                          fmaf(sW1[c * 3 + 1], y,
                          fmaf(sW1[c * 3 + 2], z, sb1[c])));
                sh1[pt][c] = (_Float16)fmaxf(h, 0.0f);
            }
        }
        __syncthreads();
        // layer2 via WMMA: this wave owns output channels [n0a, n0a+32)
        for (int mt = 0; mt < 8; ++mt) {
            v8f c0 = {}, c1 = {};
#pragma unroll
            for (int ks = 0; ks < 4; ++ks) {
                v16h a = load_a_frag(&sh1[0][0], 136, mt * 16, ks * 32);
                c0 = wmma_f16(a, load_b_frag(sW2, 128, n0a, ks * 32), c0);
                c1 = wmma_f16(a, load_b_frag(sW2, 128, n0b, ks * 32), c1);
            }
            float m0 = c0[0], m1 = c1[0];
#pragma unroll
            for (int r = 1; r < 8; ++r) { m0 = fmaxf(m0, c0[r]); m1 = fmaxf(m1, c1[r]); }
            m0 = fmaxf(m0, __shfl_xor(m0, 16));
            m1 = fmaxf(m1, __shfl_xor(m1, 16));
            rmax0 = fmaxf(rmax0, m0);
            rmax1 = fmaxf(rmax1, m1);
        }
    }
    if (lane < 16) {
        feat[(size_t)pa * 256 + n0a + lane] = rmax0 + b2[n0a + lane];
        feat[(size_t)pa * 256 + n0b + lane] = rmax1 + b2[n0b + lane];
    }
}

// ---------------- stage 2: cf = concat(feat, broadcast feat_g) ----------------
__global__ __launch_bounds__(256)
void build_cf(const float* __restrict__ feat, const float* __restrict__ feat_g,
              float* __restrict__ cf32, _Float16* __restrict__ cf16) {
    int i = blockIdx.x * 256 + threadIdx.x;  // 1024*768
    int m = i / 768, j = i - m * 768;
    float v = (j < 256) ? feat[(size_t)m * 256 + j]
                        : feat_g[(size_t)(m >> 7) * 512 + (j - 256)];
    cf32[i] = v;
    cf16[i] = (_Float16)v;
}

// ---------------- generic WMMA GEMM: C[M,N] = A[M,K] * Bw[N,K]^T + bias ----------------
// block tile 128x128, 8 waves (4x2), each wave 32x64 (2x4 WMMA tiles), K step 32.
// Double-buffered LDS tiles filled by async global->LDS copies (ASYNCcnt).
#define GF_RELU   1
#define GF_RESID  2
#define GF_ANCHOR 4
__global__ __launch_bounds__(256)
void gemm_wmma(const _Float16* __restrict__ A, const _Float16* __restrict__ Bw,
               const float* __restrict__ bias, const float* __restrict__ resid,
               const float* __restrict__ anchors, float* __restrict__ outF32,
               _Float16* __restrict__ outF16, int M, int N, int K, int flags) {
    __shared__ _Float16 sA[2][128][40];
    __shared__ _Float16 sB[2][128][40];
    const int tid = threadIdx.x, lane = tid & 31, wave = tid >> 5;
    const int wm = wave & 3, wn = wave >> 2;
    const int bm = blockIdx.x * 128, bn = blockIdx.y * 128;
    (void)M;

    // each thread async-copies 2x16B of A and 2x16B of B per K-step tile
    const int row0 = tid >> 2;                 // rows 0..63   (idx = tid)
    const int row1 = (tid + 256) >> 2;         // rows 64..127 (idx = tid+256)
    const int seg  = (tid & 3) << 3;           // 0,8,16,24 halves

    auto issue_tile = [&](int buf, int k0) {
        async_copy_b128(&sA[buf][row0][seg], A + (size_t)(bm + row0) * K + k0 + seg);
        async_copy_b128(&sA[buf][row1][seg], A + (size_t)(bm + row1) * K + k0 + seg);
        async_copy_b128(&sB[buf][row0][seg], Bw + (size_t)(bn + row0) * K + k0 + seg);
        async_copy_b128(&sB[buf][row1][seg], Bw + (size_t)(bn + row1) * K + k0 + seg);
    };

    v8f acc[2][4] = {};
    const int nk = K >> 5;
    issue_tile(0, 0);
    for (int kt = 0; kt < nk; ++kt) {
        wait_asynccnt0();       // my async copies for buf[kt&1] landed in LDS
        __syncthreads();        // everyone's copies landed
        if (kt + 1 < nk) issue_tile((kt + 1) & 1, (kt + 1) << 5);
        const _Float16* pa = &sA[kt & 1][0][0];
        const _Float16* pb = &sB[kt & 1][0][0];
        v16h a0 = load_a_frag(pa, 40, wm * 32, 0);
        v16h a1 = load_a_frag(pa, 40, wm * 32 + 16, 0);
        v16h b0 = load_b_frag(pb, 40, wn * 64, 0);
        v16h b1 = load_b_frag(pb, 40, wn * 64 + 16, 0);
        v16h b2 = load_b_frag(pb, 40, wn * 64 + 32, 0);
        v16h b3 = load_b_frag(pb, 40, wn * 64 + 48, 0);
        acc[0][0] = wmma_f16(a0, b0, acc[0][0]);
        acc[0][1] = wmma_f16(a0, b1, acc[0][1]);
        acc[0][2] = wmma_f16(a0, b2, acc[0][2]);
        acc[0][3] = wmma_f16(a0, b3, acc[0][3]);
        acc[1][0] = wmma_f16(a1, b0, acc[1][0]);
        acc[1][1] = wmma_f16(a1, b1, acc[1][1]);
        acc[1][2] = wmma_f16(a1, b2, acc[1][2]);
        acc[1][3] = wmma_f16(a1, b3, acc[1][3]);
        __syncthreads();        // all reads of buf[kt&1] done before it is refilled
    }

    const int nlo = lane & 15, hi = lane >> 4;
#pragma unroll
    for (int ti = 0; ti < 2; ++ti)
#pragma unroll
        for (int tj = 0; tj < 4; ++tj) {
            const int n = bn + wn * 64 + tj * 16 + nlo;
#pragma unroll
            for (int r = 0; r < 8; ++r) {
                const int m = bm + wm * 32 + ti * 16 + r + 8 * hi;
                float v = acc[ti][tj][r] + bias[n];
                if (flags & GF_RESID)  v += resid[(size_t)m * N + n];
                if (flags & GF_RELU)   v = fmaxf(v, 0.0f);
                if (flags & GF_ANCHOR) v += anchors[m * 3 + (n % 3)];
                if (outF32) outF32[(size_t)m * N + n] = v;
                if (outF16) outF16[(size_t)m * N + n] = (_Float16)v;
            }
        }
}

// ---------------- stage 4: attention, one block per (b, head) ----------------
__global__ __launch_bounds__(256)
void attention_wmma(const _Float16* __restrict__ qkv, _Float16* __restrict__ aout) {
    __shared__ _Float16 sK[128][200];       // key rows [a][d]
    __shared__ _Float16 sVt[192][136];      // V transposed [d][a]
    __shared__ _Float16 sP[8][16][136];     // per-wave softmax probs (f16)
    const int tid = threadIdx.x, lane = tid & 31, wave = tid >> 5;
    const int b = blockIdx.x >> 2, h = blockIdx.x & 3;
    const _Float16* base = qkv + (size_t)b * 128 * 2304 + h * 192;

    for (int idx = tid; idx < 128 * 24; idx += 256) {
        const int row = idx / 24, cc = idx - row * 24;
        *(uint4*)&sK[row][cc * 8] =
            *(const uint4*)&base[768 + (size_t)row * 2304 + cc * 8];
    }
    for (int idx = tid; idx < 128 * 192; idx += 256) {
        const int row = idx / 192, d = idx - row * 192;
        sVt[d][row] = base[1536 + (size_t)row * 2304 + d];
    }
    __syncthreads();

    const int m0 = wave * 16;
    v16h aq[6];
#pragma unroll
    for (int ks = 0; ks < 6; ++ks) aq[ks] = load_a_frag(base, 2304, m0, ks * 32);

    v8f st[8];
#pragma unroll
    for (int nt = 0; nt < 8; ++nt) {
        v8f c = {};
#pragma unroll
        for (int ks = 0; ks < 6; ++ks)
            c = wmma_f16(aq[ks], load_b_frag(&sK[0][0], 200, nt * 16, ks * 32), c);
        st[nt] = c;
    }

    const float sc = 0.072168784f;  // 1/sqrt(192)
    const int nlo = lane & 15, hi = lane >> 4;
#pragma unroll
    for (int r = 0; r < 8; ++r) {
        float mx = -3.0e38f;
#pragma unroll
        for (int nt = 0; nt < 8; ++nt) mx = fmaxf(mx, st[nt][r]);
        mx = fmaxf(mx, __shfl_xor(mx, 1));
        mx = fmaxf(mx, __shfl_xor(mx, 2));
        mx = fmaxf(mx, __shfl_xor(mx, 4));
        mx = fmaxf(mx, __shfl_xor(mx, 8));
        float sum = 0.0f;
#pragma unroll
        for (int nt = 0; nt < 8; ++nt) {
            float e = __expf((st[nt][r] - mx) * sc);
            st[nt][r] = e;
            sum += e;
        }
        sum += __shfl_xor(sum, 1);
        sum += __shfl_xor(sum, 2);
        sum += __shfl_xor(sum, 4);
        sum += __shfl_xor(sum, 8);
        const float inv = 1.0f / sum;
#pragma unroll
        for (int nt = 0; nt < 8; ++nt)
            sP[wave][r + 8 * hi][nt * 16 + nlo] = (_Float16)(st[nt][r] * inv);
    }
    __syncthreads();

    v16h ap[4];
#pragma unroll
    for (int ks = 0; ks < 4; ++ks) ap[ks] = load_a_frag(&sP[wave][0][0], 136, 0, ks * 32);

    _Float16* outb = aout + (size_t)b * 128 * 768 + h * 192;
#pragma unroll 4
    for (int dt = 0; dt < 12; ++dt) {
        v8f c = {};
#pragma unroll
        for (int ks = 0; ks < 4; ++ks)
            c = wmma_f16(ap[ks], load_b_frag(&sVt[0][0], 136, dt * 16, ks * 32), c);
#pragma unroll
        for (int r = 0; r < 8; ++r) {
            const int m = m0 + r + 8 * hi;
            outb[(size_t)m * 768 + dt * 16 + nlo] = (_Float16)c[r];
        }
    }
}

extern "C" void kernel_launch(void* const* d_in, const int* in_sizes, int n_in,
                              void* d_out, int out_size, void* d_ws, size_t ws_size,
                              hipStream_t stream) {
    (void)in_sizes; (void)n_in; (void)out_size; (void)ws_size;
    const float* regions = (const float*)d_in[0];
    const float* feat_g  = (const float*)d_in[1];
    const float* anchors = (const float*)d_in[2];
    const float* W1   = (const float*)d_in[3];
    const float* b1   = (const float*)d_in[4];
    const float* W2   = (const float*)d_in[5];
    const float* b2   = (const float*)d_in[6];
    const float* Wqkv = (const float*)d_in[7];
    const float* bqkv = (const float*)d_in[8];
    const float* Wo   = (const float*)d_in[9];
    const float* bo   = (const float*)d_in[10];
    const float* W3   = (const float*)d_in[11];
    const float* b3   = (const float*)d_in[12];
    const float* W4   = (const float*)d_in[13];
    const float* b4   = (const float*)d_in[14];

    size_t off = 0;
    auto carve = [&](size_t bytes) -> void* {
        void* p = (char*)d_ws + off;
        off += (bytes + 255) & ~(size_t)255;
        return p;
    };
    _Float16* Wqkv_h = (_Float16*)carve(2304ull * 768 * 2);
    _Float16* Wo_h   = (_Float16*)carve(768ull * 768 * 2);
    _Float16* W3_h   = (_Float16*)carve(1024ull * 768 * 2);
    _Float16* W4_h   = (_Float16*)carve(3072ull * 1024 * 2);
    float*    feat   = (float*)carve(1024ull * 256 * 4);
    float*    cf32   = (float*)carve(1024ull * 768 * 4);
    _Float16* cf16   = (_Float16*)carve(1024ull * 768 * 2);
    _Float16* qkvh   = (_Float16*)carve(1024ull * 2304 * 2);
    _Float16* aouth  = (_Float16*)carve(1024ull * 768 * 2);
    _Float16* mixedh = (_Float16*)carve(1024ull * 768 * 2);
    _Float16* mh     = (_Float16*)carve(1024ull * 1024 * 2);

    dim3 blk(256);
    // weight conversion
    {
        int n = 2304 * 768;
        cvt_f16_kernel<<<(n + 255) / 256, blk, 0, stream>>>(Wqkv, Wqkv_h, n);
        n = 768 * 768;
        cvt_f16_kernel<<<(n + 255) / 256, blk, 0, stream>>>(Wo, Wo_h, n);
        n = 1024 * 768;
        cvt_f16_kernel<<<(n + 255) / 256, blk, 0, stream>>>(W3, W3_h, n);
        n = 3072 * 1024;
        cvt_f16_kernel<<<(n + 255) / 256, blk, 0, stream>>>(W4, W4_h, n);
    }
    // stage 1: fused pointnet + maxpool
    pointnet_wmma<<<1024, blk, 0, stream>>>(regions, anchors, W1, b1, W2, b2, feat);
    // stage 2: concat
    build_cf<<<(1024 * 768) / 256, blk, 0, stream>>>(feat, feat_g, cf32, cf16);
    // stage 3: qkv projection [1024,768] x [2304,768]^T
    gemm_wmma<<<dim3(8, 18), blk, 0, stream>>>(cf16, Wqkv_h, bqkv, nullptr, nullptr,
                                               nullptr, qkvh, 1024, 2304, 768, 0);
    // stage 4: attention per (b, head)
    attention_wmma<<<32, blk, 0, stream>>>(qkvh, aouth);
    // stage 5: out-proj + residual
    gemm_wmma<<<dim3(8, 6), blk, 0, stream>>>(aouth, Wo_h, bo, cf32, nullptr,
                                              nullptr, mixedh, 1024, 768, 768, GF_RESID);
    // stage 6: mlp1 + relu
    gemm_wmma<<<dim3(8, 8), blk, 0, stream>>>(mixedh, W3_h, b3, nullptr, nullptr,
                                              nullptr, mh, 1024, 1024, 768, GF_RELU);
    // stage 7: mlp2 + anchor add -> d_out (f32)
    gemm_wmma<<<dim3(8, 24), blk, 0, stream>>>(mh, W4_h, b4, nullptr, anchors,
                                               (float*)d_out, nullptr, 1024, 3072, 1024,
                                               GF_ANCHOR);
}
